// BlockSparseMLP_49005576847766
// MI455X (gfx1250) — matrix operand order repaired and checked by reading, not compiled, not run on hardware
//
#include <hip/hip_runtime.h>

// ---------------------------------------------------------------------------
// MoE block-sparse MLP for MI455X (gfx1250, wave32, WMMA).
// zero(counts) -> router(top2 + compaction) -> prefix ->
//   gate/up GEMM + silu*mul (bf16 WMMA) -> down GEMM -> weighted combine.
// Double-buffered LDS staging (stage k+1 under compute k), b128 fragment
// loads, fragment-preload before the WMMA chain, CDNA5 async global<->LDS
// copies (ASYNCcnt), deterministic atomic-free combine.
// ---------------------------------------------------------------------------

#define TT 4096
#define HH 1024
#define FF 2816
#define EE 8

typedef __attribute__((ext_vector_type(16))) __bf16 v16bf;
typedef __attribute__((ext_vector_type(8)))  __bf16 v8bf;
typedef __attribute__((ext_vector_type(8)))  float  v8f;
typedef int v4i_vs __attribute__((vector_size(16)));   // builtin param type

union V16 { v16bf v; v8bf h[2]; };

// ---- optional CDNA5 async global<->LDS copies (ASYNCcnt) -------------------
#if defined(__HIP_DEVICE_COMPILE__) && \
    __has_builtin(__builtin_amdgcn_global_load_async_to_lds_b128) && \
    __has_builtin(__builtin_amdgcn_s_wait_asynccnt)
#define USE_ASYNC_LOAD 1
#else
#define USE_ASYNC_LOAD 0
#endif
#if defined(__HIP_DEVICE_COMPILE__) && \
    __has_builtin(__builtin_amdgcn_global_store_async_from_lds_b128) && \
    __has_builtin(__builtin_amdgcn_s_wait_asynccnt)
#define USE_ASYNC_STORE 1
#else
#define USE_ASYNC_STORE 0
#endif

#if USE_ASYNC_LOAD
__device__ __forceinline__ void async_g2l_b128(const void* g, void* l) {
    __builtin_amdgcn_global_load_async_to_lds_b128(
        (__attribute__((address_space(1))) v4i_vs*)g,
        (__attribute__((address_space(3))) v4i_vs*)l, 0, 0);
}
#endif
#if USE_ASYNC_STORE
__device__ __forceinline__ void async_l2g_b128(void* g, const void* l) {
    __builtin_amdgcn_global_store_async_from_lds_b128(
        (__attribute__((address_space(1))) v4i_vs*)g,
        (__attribute__((address_space(3))) v4i_vs*)l, 0, 0);
}
#endif
#if USE_ASYNC_LOAD || USE_ASYNC_STORE
__device__ __forceinline__ void async_wait0() {
    __builtin_amdgcn_s_wait_asynccnt(0);
}
#endif

// ---------------------------------------------------------------- zero init
__global__ void moe_zero_kernel(int* __restrict__ counts) {
    if (threadIdx.x < EE) counts[threadIdx.x] = 0;
}

// ---------------------------------------------------------------- router
__global__ void moe_router_kernel(const float* __restrict__ x,
                                  const float* __restrict__ wr,
                                  int* __restrict__ counts,
                                  int* __restrict__ idxl,
                                  int* __restrict__ sExp,
                                  int* __restrict__ sPos,
                                  float* __restrict__ sW) {
    const int lane  = threadIdx.x & 31;
    const int token = blockIdx.x * 8 + (threadIdx.x >> 5);
    float acc[EE];
#pragma unroll
    for (int e = 0; e < EE; ++e) acc[e] = 0.0f;

    const float* xr = x + (size_t)token * HH;
    for (int h = lane; h < HH; h += 32) {
        float xv = xr[h];
        const float* wrow = wr + (size_t)h * EE;
#pragma unroll
        for (int e = 0; e < EE; ++e) acc[e] = fmaf(xv, wrow[e], acc[e]);
    }
#pragma unroll
    for (int off = 16; off > 0; off >>= 1) {
#pragma unroll
        for (int e = 0; e < EE; ++e) acc[e] += __shfl_xor(acc[e], off, 32);
    }
    if (lane == 0) {
        int i1 = 0; float l1 = acc[0];
#pragma unroll
        for (int e = 1; e < EE; ++e)
            if (acc[e] > l1) { l1 = acc[e]; i1 = e; }
        int i2 = -1; float l2 = -3.4e38f;
#pragma unroll
        for (int e = 0; e < EE; ++e)
            if (e != i1 && acc[e] > l2) { l2 = acc[e]; i2 = e; }
        float w1 = 1.0f / (1.0f + __expf(l2 - l1));   // top-2 softmax renorm
        float w2 = 1.0f - w1;
        int p1 = atomicAdd(&counts[i1], 1);
        idxl[i1 * TT + p1] = token;
        sExp[2 * token] = i1; sPos[2 * token] = p1; sW[2 * token] = w1;
        int p2 = atomicAdd(&counts[i2], 1);
        idxl[i2 * TT + p2] = token;
        sExp[2 * token + 1] = i2; sPos[2 * token + 1] = p2; sW[2 * token + 1] = w2;
    }
}

// ---------------------------------------------------------------- prefix sum
__global__ void moe_prefix_kernel(const int* __restrict__ counts,
                                  int* __restrict__ offsets) {
    if (threadIdx.x == 0 && blockIdx.x == 0) {
        int s = 0;
#pragma unroll
        for (int e = 0; e < EE; ++e) { offsets[e] = s; s += counts[e]; }
    }
}

// Stage a 32(K) x 128(N) fp32 tile into LDS in *fragment order*
// [nsub(8)][kg(2)][nl(16)][j(16)] so consuming lanes read 2x b128 each.
__device__ __forceinline__ void stage_b_tile(const float* __restrict__ src,
                                             size_t rowStride,
                                             __bf16* __restrict__ sB,
                                             int tid, bool pref) {
    const int m  = tid >> 4;          // k-pair index 0..15
    const int n0 = (tid & 15) * 8;    // col base 0..120
    const float* r0 = src + (size_t)(2 * m) * rowStride + n0;
    const float* r1 = r0 + rowStride;
    if (pref) __builtin_prefetch(r0 + 64 * rowStride, 0, 1);
    float4 a0 = *(const float4*)r0;
    float4 a1 = *(const float4*)(r0 + 4);
    float4 b0 = *(const float4*)r1;
    float4 b1 = *(const float4*)(r1 + 4);
    float fa[8] = {a0.x, a0.y, a0.z, a0.w, a1.x, a1.y, a1.z, a1.w};
    float fb[8] = {b0.x, b0.y, b0.z, b0.w, b1.x, b1.y, b1.z, b1.w};
    const int kg2 = m >> 3;
    const int j   = (2 * m) & 15;     // even -> 4B-aligned pair store
#pragma unroll
    for (int i = 0; i < 8; ++i) {
        int n = n0 + i;
        int nsub = n >> 4, nl = n & 15;
        union { unsigned u; __bf16 hh[2]; } p;
        p.hh[0] = (__bf16)fa[i];
        p.hh[1] = (__bf16)fb[i];
        *(unsigned*)&sB[(((nsub * 2 + kg2) * 16 + nl) * 16 + j)] = p.u;
    }
}

// ---------------------------------------------------------------- gate/up
__global__ void moe_gateup_kernel(const float* __restrict__ x,
                                  const float* __restrict__ wg,
                                  const float* __restrict__ wu,
                                  const int* __restrict__ counts,
                                  const int* __restrict__ offsets,
                                  const int* __restrict__ idxl,
                                  __bf16* __restrict__ act) {
    const int e = blockIdx.z;
    const int cnt = counts[e];
    const int tileBase = blockIdx.y * 64;
    if (tileBase >= cnt) return;
    const int fBase = blockIdx.x * 128;

    __shared__ __bf16 sA[2][64 * 32];
    __shared__ __bf16 sBg[2][32 * 128];
    __shared__ __bf16 sBu[2][32 * 128];
    __shared__ __bf16 sOut[64 * 128];
    __shared__ int    sTok[64];

    const int tid  = threadIdx.x;
    const int lane = tid & 31;
    const int wv   = tid >> 5;
    const int mb   = wv & 3;   // 16-row block
    const int nh   = wv >> 2;  // 64-col half
    const int kg   = lane >> 4;
    const int nl   = lane & 15;

    if (tid < 64) {
        int slot = tileBase + tid;
        sTok[tid] = (slot < cnt) ? idxl[e * TT + slot] : 0;
    }
    __syncthreads();

    const size_t wbase = (size_t)e * HH * FF;
    v8f accg[4], accu[4];
#pragma unroll
    for (int nb = 0; nb < 4; ++nb) { accg[nb] = {}; accu[nb] = {}; }

    const int ar = tid >> 2;            // A staging: row 0..63
    const int ak = (tid & 3) * 8;       //            k  0,8,16,24
    const size_t axrow = (size_t)sTok[ar] * HH + ak;

    // stage helper for A tile (fp32 gather -> bf16 -> b128 store)
    auto stageA = [&](int buf, int k0) {
        const float* src = x + axrow + k0;
        float4 f0 = *(const float4*)src;
        float4 f1 = *(const float4*)(src + 4);
        v8bf av;
        av[0] = (__bf16)f0.x; av[1] = (__bf16)f0.y;
        av[2] = (__bf16)f0.z; av[3] = (__bf16)f0.w;
        av[4] = (__bf16)f1.x; av[5] = (__bf16)f1.y;
        av[6] = (__bf16)f1.z; av[7] = (__bf16)f1.w;
        *(v8bf*)&sA[buf][ar * 32 + ak] = av;
    };

    // prologue: stage k0 = 0 into buffer 0
    stageA(0, 0);
    stage_b_tile(wg + wbase + fBase, FF, sBg[0], tid, true);
    stage_b_tile(wu + wbase + fBase, FF, sBu[0], tid, true);
    __syncthreads();

    int buf = 0;
    for (int k0 = 0; k0 < HH; k0 += 32) {
        const int nxt = buf ^ 1;
        const bool more = (k0 + 32 < HH);
        if (more) {  // stage next K-step under this step's compute
            stageA(nxt, k0 + 32);
            stage_b_tile(wg + wbase + (size_t)(k0 + 32) * FF + fBase, FF,
                         sBg[nxt], tid, k0 + 64 < HH);
            stage_b_tile(wu + wbase + (size_t)(k0 + 32) * FF + fBase, FF,
                         sBu[nxt], tid, k0 + 64 < HH);
        }
        // preload all fragments, then run the WMMA chain
        V16 af;
        {
            const __bf16* ap = &sA[buf][(mb * 16 + nl) * 32 + kg * 16];
            af.h[0] = *(const v8bf*)ap;
            af.h[1] = *(const v8bf*)(ap + 8);
        }
        V16 bgf[4], buf_[4];
#pragma unroll
        for (int nb = 0; nb < 4; ++nb) {
            const int bb = (((nh * 4 + nb) * 2 + kg) * 16 + nl) * 16;
            bgf[nb].h[0]  = *(const v8bf*)&sBg[buf][bb];
            bgf[nb].h[1]  = *(const v8bf*)&sBg[buf][bb + 8];
            buf_[nb].h[0] = *(const v8bf*)&sBu[buf][bb];
            buf_[nb].h[1] = *(const v8bf*)&sBu[buf][bb + 8];
        }
#pragma unroll
        for (int nb = 0; nb < 4; ++nb) {
            accg[nb] = __builtin_amdgcn_wmma_f32_16x16x32_bf16(
                false, af.v, false, bgf[nb].v, (short)0, accg[nb], false, false);
            accu[nb] = __builtin_amdgcn_wmma_f32_16x16x32_bf16(
                false, af.v, false, buf_[nb].v, (short)0, accu[nb], false, false);
        }
        __syncthreads();
        buf = nxt;
    }

    // epilogue: silu(g)*u -> LDS transpose -> coalesced b128 global stores
#pragma unroll
    for (int nb = 0; nb < 4; ++nb) {
        const int c = nh * 64 + nb * 16 + nl;
#pragma unroll
        for (int r = 0; r < 8; ++r) {
            int ml = mb * 16 + kg * 8 + r;
            float g = accg[nb][r];
            float u = accu[nb][r];
            sOut[ml * 128 + c] = (__bf16)(g / (1.0f + __expf(-g)) * u);
        }
    }
    __syncthreads();
    {
        const int off = offsets[e];
        const int r  = tid >> 2;
        const int c0 = (tid & 3) * 32;
        const int slot = tileBase + r;
        if (slot < cnt) {
            __bf16* dst = act + (size_t)(off + slot) * FF + fBase + c0;
#pragma unroll
            for (int i = 0; i < 4; ++i) {
#if USE_ASYNC_STORE
                async_l2g_b128(dst + i * 8, &sOut[r * 128 + c0 + i * 8]);
#else
                *(v8bf*)(dst + i * 8) = *(const v8bf*)&sOut[r * 128 + c0 + i * 8];
#endif
            }
        }
    }
    // S_ENDPGM implies wait-idle, async stores drain before kernel end.
}

// ---------------------------------------------------------------- down GEMM
__global__ void moe_down_kernel(const __bf16* __restrict__ act,
                                const float* __restrict__ wd,
                                const int* __restrict__ counts,
                                const int* __restrict__ offsets,
                                float* __restrict__ dbuf) {
    const int e = blockIdx.z;
    const int cnt = counts[e];
    const int tileBase = blockIdx.y * 64;
    if (tileBase >= cnt) return;
    const int hBase = blockIdx.x * 128;

    __shared__ __bf16 sA[2][64 * 32];
    __shared__ __bf16 sB[2][32 * 128];
    __shared__ float  sOut[64 * 128];

    const int tid  = threadIdx.x;
    const int lane = tid & 31;
    const int wv   = tid >> 5;
    const int mb   = wv & 3;
    const int nh   = wv >> 2;
    const int kg   = lane >> 4;
    const int nl   = lane & 15;

    const int off = offsets[e];
    const size_t wbase = (size_t)e * FF * HH;
    v8f acc[4];
#pragma unroll
    for (int nb = 0; nb < 4; ++nb) acc[nb] = {};

    const int ar = tid >> 2;
    const int ak = (tid & 3) * 8;
    const int aslot = tileBase + ar;
    const bool avalid = (aslot < cnt);
    const __bf16* asrc = act + (size_t)(off + aslot) * FF + ak;

    // A tile staging: pure bf16 global->LDS b128 copy (async path if present;
    // invalid rows never feed a stored result, stale LDS there is harmless).
    auto stageA = [&](int bufi, int k0) {
#if USE_ASYNC_LOAD
        if (avalid) async_g2l_b128(asrc + k0, &sA[bufi][ar * 32 + ak]);
#else
        v8bf av = {};
        if (avalid) av = *(const v8bf*)(asrc + k0);
        *(v8bf*)&sA[bufi][ar * 32 + ak] = av;
#endif
    };

    stageA(0, 0);
    stage_b_tile(wd + wbase + hBase, HH, sB[0], tid, true);
#if USE_ASYNC_LOAD
    async_wait0();
#endif
    __syncthreads();

    int buf = 0;
    for (int k0 = 0; k0 < FF; k0 += 32) {
        const int nxt = buf ^ 1;
        const bool more = (k0 + 32 < FF);
        if (more) {
            stageA(nxt, k0 + 32);
            stage_b_tile(wd + wbase + (size_t)(k0 + 32) * HH + hBase, HH,
                         sB[nxt], tid, k0 + 64 < FF);
        }
        V16 af;
        {
            const __bf16* ap = &sA[buf][(mb * 16 + nl) * 32 + kg * 16];
            af.h[0] = *(const v8bf*)ap;
            af.h[1] = *(const v8bf*)(ap + 8);
        }
        V16 bf_[4];
#pragma unroll
        for (int nb = 0; nb < 4; ++nb) {
            const int bb = (((nh * 4 + nb) * 2 + kg) * 16 + nl) * 16;
            bf_[nb].h[0] = *(const v8bf*)&sB[buf][bb];
            bf_[nb].h[1] = *(const v8bf*)&sB[buf][bb + 8];
        }
#pragma unroll
        for (int nb = 0; nb < 4; ++nb) {
            acc[nb] = __builtin_amdgcn_wmma_f32_16x16x32_bf16(
                false, af.v, false, bf_[nb].v, (short)0, acc[nb], false, false);
        }
#if USE_ASYNC_LOAD
        if (more) async_wait0();   // own async A copy must land before barrier
#endif
        __syncthreads();
        buf = nxt;
    }

    // epilogue: LDS transpose -> coalesced float4 stores of raw d rows
#pragma unroll
    for (int nb = 0; nb < 4; ++nb) {
        const int c = nh * 64 + nb * 16 + nl;
#pragma unroll
        for (int r = 0; r < 8; ++r) {
            int ml = mb * 16 + kg * 8 + r;
            sOut[ml * 128 + c] = acc[nb][r];
        }
    }
    __syncthreads();
    {
        const int r  = tid >> 2;
        const int c0 = (tid & 3) * 32;
        const int slot = tileBase + r;
        if (slot < cnt) {
            float* dst = dbuf + (size_t)(off + slot) * HH + hBase + c0;
#pragma unroll
            for (int i = 0; i < 8; i += 2) {
#if USE_ASYNC_STORE
                async_l2g_b128(dst + i * 4, &sOut[r * 128 + c0 + i * 4]);
                async_l2g_b128(dst + i * 4 + 4, &sOut[r * 128 + c0 + i * 4 + 4]);
#else
                *(float4*)(dst + i * 4)     = *(const float4*)&sOut[r * 128 + c0 + i * 4];
                *(float4*)(dst + i * 4 + 4) = *(const float4*)&sOut[r * 128 + c0 + i * 4 + 4];
#endif
            }
        }
    }
}

// ---------------------------------------------------------------- combine
// out[t] = w1 * d[row(t,0)] + w2 * d[row(t,1)]  (float4, coalesced, exactly
// two commutative fp32 adds per element -> deterministic).
__global__ void moe_combine_kernel(const float* __restrict__ dbuf,
                                   const int* __restrict__ offsets,
                                   const int* __restrict__ sExp,
                                   const int* __restrict__ sPos,
                                   const float* __restrict__ sW,
                                   float* __restrict__ out) {
    const int gid = blockIdx.x * blockDim.x + threadIdx.x; // T*H/4 threads
    const int t = gid >> 8;            // H/4 = 256 float4 per token
    const int c = (gid & 255) * 4;
    const int e1 = sExp[2 * t], e2 = sExp[2 * t + 1];
    const int r1 = offsets[e1] + sPos[2 * t];
    const int r2 = offsets[e2] + sPos[2 * t + 1];
    const float w1 = sW[2 * t], w2 = sW[2 * t + 1];
    float4 d1 = *(const float4*)&dbuf[(size_t)r1 * HH + c];
    float4 d2 = *(const float4*)&dbuf[(size_t)r2 * HH + c];
    float4 o;
    o.x = w1 * d1.x + w2 * d2.x;
    o.y = w1 * d1.y + w2 * d2.y;
    o.z = w1 * d1.z + w2 * d2.z;
    o.w = w1 * d1.w + w2 * d2.w;
    *(float4*)&out[(size_t)t * HH + c] = o;
}

// ---------------------------------------------------------------- launch
extern "C" void kernel_launch(void* const* d_in, const int* in_sizes, int n_in,
                              void* d_out, int out_size, void* d_ws, size_t ws_size,
                              hipStream_t stream) {
    const float* x  = (const float*)d_in[0];
    const float* wr = (const float*)d_in[1];
    const float* wg = (const float*)d_in[2];
    const float* wu = (const float*)d_in[3];
    const float* wd = (const float*)d_in[4];
    float* out = (float*)d_out;

    // workspace carve-up (~80 MB)
    char* w = (char*)d_ws;
    int*    counts  = (int*)w;    w += 256;
    int*    offsets = (int*)w;    w += 256;
    int*    idxl    = (int*)w;    w += (size_t)EE * TT * sizeof(int);
    int*    sExp    = (int*)w;    w += (size_t)2 * TT * sizeof(int);
    int*    sPos    = (int*)w;    w += (size_t)2 * TT * sizeof(int);
    float*  sW      = (float*)w;  w += (size_t)2 * TT * sizeof(float);
    __bf16* actbuf  = (__bf16*)w; w += (size_t)2 * TT * FF * sizeof(__bf16);
    float*  dbuf    = (float*)w;  // 2T x H fp32 = 33.5 MB

    moe_zero_kernel<<<1, 32, 0, stream>>>(counts);
    moe_router_kernel<<<TT / 8, 256, 0, stream>>>(x, wr, counts, idxl,
                                                  sExp, sPos, sW);
    moe_prefix_kernel<<<1, 1, 0, stream>>>(counts, offsets);
    moe_gateup_kernel<<<dim3(FF / 128, TT / 64, EE), 256, 0, stream>>>(
        x, wg, wu, counts, offsets, idxl, actbuf);
    moe_down_kernel<<<dim3(HH / 128, TT / 64, EE), 256, 0, stream>>>(
        actbuf, wd, counts, offsets, dbuf);
    moe_combine_kernel<<<(TT * HH / 4) / 256, 256, 0, stream>>>(
        dbuf, offsets, sExp, sPos, sW, out);
}
